// Attention_17806934409481
// MI455X (gfx1250) — matrix operand order repaired and checked by reading, not compiled, not run on hardware
//
#include <hip/hip_runtime.h>
#include <hip/hip_bf16.h>

// ---------------------------------------------------------------------------
// Problem constants (match reference): B=2, S=2048, H=1024, N=16, D=64
// ---------------------------------------------------------------------------
static constexpr int Bb = 2;
static constexpr int Ss = 2048;
static constexpr int Hh = 1024;
static constexpr int Nn = 16;
static constexpr int Dd = 64;
static constexpr int NDh = Nn * Dd;      // 1024
static constexpr int QKV3 = 3 * NDh;     // 3072
static constexpr float SCALE = 0.125f;   // D^-0.5
static constexpr float PAD = -30000.0f;

typedef __attribute__((ext_vector_type(16))) __bf16 bf16x16;
typedef __attribute__((ext_vector_type(8)))  __bf16 bf16x8;
typedef __attribute__((ext_vector_type(2)))  __bf16 bf16x2;
typedef __attribute__((ext_vector_type(8)))  float  f32x8;
typedef __attribute__((ext_vector_type(4)))  unsigned int uint32x4;
typedef __attribute__((ext_vector_type(8)))  int    int32x8;
typedef __attribute__((ext_vector_type(4)))  int    int32x4;

#define WMMA_BF16(a, b, c) \
    __builtin_amdgcn_wmma_f32_16x16x32_bf16(false, (a), false, (b), (short)0, (c), false, false)

#if __has_builtin(__builtin_amdgcn_tensor_load_to_lds) && \
    __has_builtin(__builtin_amdgcn_s_wait_tensorcnt)
#define HAVE_TDM 1
#else
#define HAVE_TDM 0
#endif

// ---------------------------------------------------------------------------
// Fragment loaders (WMMA 16x16x32 bf16, wave32; this file's contiguous k order, same for A and B)
// A (16xK=32): lane holds row r=lane&15; K elems = ko..ko+15,
//              ko = (lane>>4)*16.  Source: row-major A, leading dim ld.
// B (K=32x16): lane holds col c=lane&15; K elems = kb..kb+15, kb=(lane>>4)*16.
//              Source: row-major B^T (i.e. [16 cols][32 K]), leading dim ld.
// ---------------------------------------------------------------------------
__device__ __forceinline__ bf16x16 frag_a(const __bf16* p, int ld, int lane) {
    const __bf16* q = p + (lane & 15) * ld + ((lane >> 4) << 4);
    bf16x8 lo = *(const bf16x8*)(q);
    bf16x8 hi = *(const bf16x8*)(q + 8);
    bf16x16 r;
#pragma unroll
    for (int i = 0; i < 8; ++i) { r[i] = lo[i]; r[8 + i] = hi[i]; }
    return r;
}

__device__ __forceinline__ bf16x16 frag_b(const __bf16* pT, int ld, int lane) {
    const __bf16* q = pT + (lane & 15) * ld + ((lane >> 4) << 4);
    bf16x8 lo = *(const bf16x8*)(q);
    bf16x8 hi = *(const bf16x8*)(q + 8);
    bf16x16 r;
#pragma unroll
    for (int i = 0; i < 8; ++i) { r[i] = lo[i]; r[8 + i] = hi[i]; }
    return r;
}

__device__ __forceinline__ void st_pair(__bf16* dst, float a, float b) {
    bf16x2 v; v[0] = (__bf16)a; v[1] = (__bf16)b;
    *(bf16x2*)dst = v;                       // one ds_store_b32
}

// ---------------------------------------------------------------------------
// TDM: issue tensor_load_to_lds of a [rows x 32] bf16 tile (row stride ldg
// elements in global) into LDS at lds_addr, with HW padding of 4 DWORDs every
// 16 DWORDs -> LDS row stride 40 bf16 (matches frag loaders).
// ---------------------------------------------------------------------------
#if HAVE_TDM
__device__ __forceinline__ void tdm_load_tile_b16(const __bf16* gsrc,
                                                  unsigned lds_addr,
                                                  int rows, int ldg,
                                                  int tdim1) {
    unsigned long long ga = (unsigned long long)(size_t)gsrc;
    uint32x4 g0;
    g0[0] = 1u;                                           // count=1 (valid)
    g0[1] = lds_addr;                                     // LDS byte address
    g0[2] = (unsigned)(ga & 0xFFFFFFFFu);
    g0[3] = (unsigned)((ga >> 32) & 0x1FFFFFFu) | (2u << 30);  // type=2 (image)

    // group1: data_size=2B(code1), pad_enable, pad_interval=16dw(code3),
    //         pad_amount=4dw(code3)
    unsigned u0 = (1u << 16) | (1u << 20) | (3u << 22) | (3u << 25);
    unsigned td0 = (unsigned)ldg;           // tensor dim0 (elements)
    unsigned td1 = (unsigned)tdim1;         // tensor dim1 (rows)
    unsigned u1 = (td0 & 0xFFFFu) << 16;                  // tensor_dim0[15:0] @ bits63:48
    unsigned u2 = (td0 >> 16) | ((td1 & 0xFFFFu) << 16);  // dim0 hi | dim1 lo
    unsigned u3 = (td1 >> 16) | (32u << 16);              // dim1 hi | tile_dim0=32
    unsigned u4 = (unsigned)rows;                         // tile_dim1 | tile_dim2=0
    unsigned u5 = (unsigned)ldg;                          // tensor_dim0_stride lo32
    unsigned u6 = 0u;                                     // stride0 hi | stride1 lo
    unsigned u7 = 0u;
    int32x8 g1 = {(int)u0, (int)u1, (int)u2, (int)u3,
                  (int)u4, (int)u5, (int)u6, (int)u7};
    int32x4 gz = {0, 0, 0, 0};
#if defined(__clang_major__) && __clang_major__ >= 23
    int32x8 gz8 = {0, 0, 0, 0, 0, 0, 0, 0};
    __builtin_amdgcn_tensor_load_to_lds(g0, g1, gz, gz, gz8, 0);
#else
    __builtin_amdgcn_tensor_load_to_lds(g0, g1, gz, gz, 0);
#endif
}
#endif

// ---------------------------------------------------------------------------
// Kernel 1: cum[n,t] = cumsum_t rpe[n,t]   (tiny: 16 x 2048)
// ---------------------------------------------------------------------------
__global__ void k_cumsum(const float* __restrict__ rpe, float* __restrict__ cum) {
    int n = blockIdx.x;
    if (threadIdx.x == 0) {
        float a = 0.f;
        for (int t = 0; t < Ss; ++t) { a += rpe[n * Ss + t]; cum[n * Ss + t] = a; }
    }
}

// ---------------------------------------------------------------------------
// Kernel 2: QKV projection.  X[4096,1024] * W[1024,3072] -> q/k/vT (bf16)
// Block: 256 thr (8 waves). Block tile 128(M) x 64(N), K-step 32 via LDS.
// ---------------------------------------------------------------------------
__global__ __launch_bounds__(256) void k_qkv(const float* __restrict__ x,
                                             const float* __restrict__ w,
                                             __bf16* __restrict__ qo,
                                             __bf16* __restrict__ ko,
                                             __bf16* __restrict__ vT) {
    __shared__ __bf16 sA[128 * 40];
    __shared__ __bf16 sB[64 * 40];
    const int tid = threadIdx.x;
    const int lane = tid & 31, wave = tid >> 5;
    const int m0 = blockIdx.x * 128;
    const int n0 = blockIdx.y * 64;

    f32x8 acc[4] = {};
    for (int k0 = 0; k0 < Hh; k0 += 32) {
        __syncthreads();
        // stage A: 128x32 f32 -> bf16, packed pair stores (ds_store_b32)
#pragma unroll
        for (int e = 0; e < 8; ++e) {
            int idx = e * 256 + tid;            // 2048 pairs
            int r = idx >> 4, cp = idx & 15;
            const float2 v = *(const float2*)&x[(m0 + r) * Hh + k0 + cp * 2];
            st_pair(&sA[r * 40 + cp * 2], v.x, v.y);
        }
        // stage B transposed: [c][k], pack along k
#pragma unroll
        for (int e = 0; e < 4; ++e) {
            int idx = e * 256 + tid;            // 1024 pairs
            int c = idx & 63, kp = idx >> 6;
            float a0 = w[(k0 + kp * 2 + 0) * QKV3 + n0 + c];
            float a1 = w[(k0 + kp * 2 + 1) * QKV3 + n0 + c];
            st_pair(&sB[c * 40 + kp * 2], a0, a1);
        }
        __syncthreads();
        bf16x16 a = frag_a(&sA[wave * 16 * 40], 40, lane);
#pragma unroll
        for (int t = 0; t < 4; ++t) {
            bf16x16 b = frag_b(&sB[t * 16 * 40], 40, lane);
            acc[t] = WMMA_BF16(a, b, acc[t]);
        }
    }
    // Epilogue: branch-free per-tile pointer select into q / k / v^T
    const int half = (lane >> 4) * 8;
    const int mbase = m0 + wave * 16 + half;          // 8 consecutive rows
    const int bb = mbase >> 11, s0 = mbase & 2047;    // uniform within tile
#pragma unroll
    for (int t = 0; t < 4; ++t) {
        int col = n0 + t * 16 + (lane & 15);
        int m3 = col >> 10;                            // 0=q,1=k,2=v (uniform/tile)
        int rem = col & 1023;
        int hn = rem >> 6, d = rem & 63;
        int bn = bb * Nn + hn;
        size_t off_qk = ((size_t)bn * Ss + s0) * Dd + d;
        size_t off_v  = ((size_t)bn * Dd + d) * Ss + s0;
        __bf16* dst = (m3 == 0) ? qo + off_qk : (m3 == 1) ? ko + off_qk : vT + off_v;
        int step = (m3 == 2) ? 1 : Dd;
#pragma unroll
        for (int i = 0; i < 8; ++i) dst[i * step] = (__bf16)acc[t][i];
    }
}

// ---------------------------------------------------------------------------
// Kernel 3: flash attention (transposed). One wave = 16 queries, full D=64.
//   S^T tile = K_tile(16x64) @ Q^T(64x16) ; online softmax per query;
//   mix^T += V^T(16x32) @ P^T(32x16) with P^T built via half-wave shfl.
// Grid: (B*N, S/64), block 128 (4 independent waves).
// ---------------------------------------------------------------------------
__global__ __launch_bounds__(128) void k_attn(const __bf16* __restrict__ q,
                                              const __bf16* __restrict__ k,
                                              const __bf16* __restrict__ vT,
                                              const float* __restrict__ cum,
                                              __bf16* __restrict__ mix) {
    const int lane = threadIdx.x & 31;
    const int wave = threadIdx.x >> 5;
    const int bn = blockIdx.x;               // 0..31
    const int bb = bn >> 4, hn = bn & 15;
    const int q0 = (blockIdx.y * 4 + wave) * 16;

    const __bf16* qm = q  + (size_t)bn * Ss * Dd;
    const __bf16* km = k  + (size_t)bn * Ss * Dd;
    const __bf16* vm = vT + (size_t)bn * Dd * Ss;
    const float*  cm = cum + hn * Ss;

    const int qcol = lane & 15;
    const int half = (lane >> 4) * 8;
    const int qg = q0 + qcol;
    const bool lo16 = (lane < 16);

    bf16x16 bq0 = frag_b(qm + q0 * Dd,      Dd, lane);
    bf16x16 bq1 = frag_b(qm + q0 * Dd + 32, Dd, lane);

    f32x8 acc[4] = {};
    float mi = -1.0e30f, li = 0.f;

    const int nkt = (q0 + 47) >> 5;          // causal: key tiles of 32
    for (int kt = 0; kt < nkt; ++kt) {
        const int key0 = kt * 32;
        if (kt + 1 < nkt) __builtin_prefetch(km + (key0 + 32) * Dd, 0, 0);

        f32x8 st0 = {}, st1 = {};
        {
            bf16x16 ak = frag_a(km + key0 * Dd, Dd, lane);
            st0 = WMMA_BF16(ak, bq0, st0);
            ak = frag_a(km + key0 * Dd + 32, Dd, lane);
            st0 = WMMA_BF16(ak, bq1, st0);
            ak = frag_a(km + (key0 + 16) * Dd, Dd, lane);
            st1 = WMMA_BF16(ak, bq0, st1);
            ak = frag_a(km + (key0 + 16) * Dd + 32, Dd, lane);
            st1 = WMMA_BF16(ak, bq1, st1);
        }

        float p0[8], p1[8];
        float mt = -1.0e30f;
#pragma unroll
        for (int i = 0; i < 8; ++i) {
            int j = key0 + half + i;
            float bsa = cm[j <= qg ? (qg - j) : 0];
            float v = st0[i] * SCALE + (j <= qg ? bsa : PAD);
            p0[i] = v; mt = fmaxf(mt, v);
            j += 16;
            float bsb = cm[j <= qg ? (qg - j) : 0];
            v = st1[i] * SCALE + (j <= qg ? bsb : PAD);
            p1[i] = v; mt = fmaxf(mt, v);
        }
        mt = fmaxf(mt, __shfl_xor(mt, 16, 32));

        const float mnew = fmaxf(mi, mt);
        const float corr = __expf(mi - mnew);

        float ls = 0.f;
        bf16x16 bp;                       // P^T B-frag: lanes0-15 keys 0-15
#pragma unroll
        for (int i = 0; i < 8; ++i) {
            float e0 = __expf(p0[i] - mnew);
            float e1 = __expf(p1[i] - mnew);
            ls += e0 + e1;
            float o0 = __shfl_xor(e0, 16, 32);
            float o1 = __shfl_xor(e1, 16, 32);
            bp[i]     = (__bf16)(lo16 ? e0 : o1);
            bp[8 + i] = (__bf16)(lo16 ? o0 : e1);
        }
        ls += __shfl_xor(ls, 16, 32);
        li = li * corr + ls;

#pragma unroll
        for (int t = 0; t < 4; ++t)
#pragma unroll
            for (int i = 0; i < 8; ++i) acc[t][i] *= corr;

#pragma unroll
        for (int t = 0; t < 4; ++t) {
            bf16x16 av = frag_a(vm + (t * 16) * Ss + key0, Ss, lane);
            acc[t] = WMMA_BF16(av, bp, acc[t]);
        }
        mi = mnew;
    }

    const float inv = 1.0f / li;
    __bf16* dst = mix + ((size_t)(bb * Ss + qg)) * NDh + hn * Dd;
#pragma unroll
    for (int t = 0; t < 4; ++t) {
        int d0 = t * 16 + half;
#pragma unroll
        for (int i = 0; i < 8; ++i) dst[d0 + i] = (__bf16)(acc[t][i] * inv);
    }
}

// ---------------------------------------------------------------------------
// Kernel 4: output projection.  mix[4096,1024](bf16) * W[1024,1024] -> out f32
// Double-buffered: A tile shipped by TDM (tensor_load_to_lds, HW-padded to
// stride 40), B tile staged+converted by threads. Overlap: issue TDM k+1,
// compute k, s_wait_tensorcnt 0, barrier.
// ---------------------------------------------------------------------------
__global__ __launch_bounds__(256) void k_out(const __bf16* __restrict__ mix,
                                             const float* __restrict__ w,
                                             float* __restrict__ out) {
    __shared__ __bf16 sA[2][128 * 40];
    __shared__ __bf16 sB[2][64 * 40];
    const int tid = threadIdx.x;
    const int lane = tid & 31, wave = tid >> 5;
    const int m0 = blockIdx.x * 128;
    const int n0 = blockIdx.y * 64;
    const int NK = NDh / 32;

    // ---- stage helpers -----------------------------------------------------
    auto stageB = [&](int buf, int k0) {
#pragma unroll
        for (int e = 0; e < 4; ++e) {
            int idx = e * 256 + tid;
            int c = idx & 63, kp = idx >> 6;
            float a0 = w[(k0 + kp * 2 + 0) * Hh + n0 + c];
            float a1 = w[(k0 + kp * 2 + 1) * Hh + n0 + c];
            st_pair(&sB[buf][c * 40 + kp * 2], a0, a1);
        }
    };
    auto stageA = [&](int buf, int k0) {
#if HAVE_TDM
        if (wave == 0) {
            tdm_load_tile_b16(mix + (size_t)m0 * NDh + k0,
                              (unsigned)(size_t)(void*)&sA[buf][0],
                              128, NDh, Bb * Ss);
        }
#else
#pragma unroll
        for (int e = 0; e < 8; ++e) {
            int idx = e * 256 + tid;
            int r = idx >> 4, cp = idx & 15;
            *(bf16x2*)&sA[buf][r * 40 + cp * 2] =
                *(const bf16x2*)&mix[(size_t)(m0 + r) * NDh + k0 + cp * 2];
        }
#endif
    };

    // ---- prologue ----------------------------------------------------------
    stageB(0, 0);
    stageA(0, 0);
#if HAVE_TDM
    if (wave == 0) __builtin_amdgcn_s_wait_tensorcnt(0);
#endif
    __syncthreads();

    f32x8 acc[4] = {};
    int cur = 0;
    for (int kb = 0; kb < NK; ++kb) {
        const int nxt = cur ^ 1;
        if (kb + 1 < NK) {                   // overlap next-tile staging
            stageB(nxt, (kb + 1) * 32);
            stageA(nxt, (kb + 1) * 32);
        }
        bf16x16 a = frag_a(&sA[cur][wave * 16 * 40], 40, lane);
#pragma unroll
        for (int t = 0; t < 4; ++t) {
            bf16x16 b = frag_b(&sB[cur][t * 16 * 40], 40, lane);
            acc[t] = WMMA_BF16(a, b, acc[t]);
        }
#if HAVE_TDM
        if (wave == 0) __builtin_amdgcn_s_wait_tensorcnt(0);
#endif
        __syncthreads();
        cur = nxt;
    }

    const int half = (lane >> 4) * 8;
#pragma unroll
    for (int t = 0; t < 4; ++t) {
        int h = n0 + t * 16 + (lane & 15);
#pragma unroll
        for (int i = 0; i < 8; ++i) {
            int m = m0 + wave * 16 + half + i;
            out[(size_t)m * Hh + h] = acc[t][i];
        }
    }
}

// ---------------------------------------------------------------------------
// Launch.  ws layout (bytes):
//   [0, 128K)     cum f32 N*S
//   [+128K..]     q bf16 [bn][s][d] | k bf16 [bn][s][d] | vT bf16 [bn][d][s]
//   [..+32M]      mix bf16 [b][s][n*d]
// ---------------------------------------------------------------------------
extern "C" void kernel_launch(void* const* d_in, const int* in_sizes, int n_in,
                              void* d_out, int out_size, void* d_ws, size_t ws_size,
                              hipStream_t stream) {
    const float* x    = (const float*)d_in[0];   // [B,S,H]
    const float* qkvw = (const float*)d_in[1];   // [H,3,N,D] = [1024,3072]
    const float* outw = (const float*)d_in[2];   // [N,D,H]   = [1024,1024]
    const float* rpe  = (const float*)d_in[3];   // [N,S]

    char* ws = (char*)d_ws;
    const size_t QKV_BYTES = (size_t)Bb * Nn * Ss * Dd * sizeof(__bf16); // 8 MiB
    float*  cum = (float*)ws;
    __bf16* qb  = (__bf16*)(ws + 131072);
    __bf16* kb  = (__bf16*)(ws + 131072 + QKV_BYTES);
    __bf16* vT  = (__bf16*)(ws + 131072 + 2 * QKV_BYTES);
    __bf16* mix = (__bf16*)(ws + 131072 + 3 * QKV_BYTES);

    k_cumsum<<<dim3(Nn), dim3(32), 0, stream>>>(rpe, cum);
    k_qkv  <<<dim3((Bb * Ss) / 128, QKV3 / 64), dim3(256), 0, stream>>>(x, qkvw, qb, kb, vT);
    k_attn <<<dim3(Bb * Nn, Ss / 64), dim3(128), 0, stream>>>(qb, kb, vT, cum, mix);
    k_out  <<<dim3((Bb * Ss) / 128, Hh / 64), dim3(256), 0, stream>>>(mix, outw, (float*)d_out);
}
